// MultiHeadAttentionBlock_74156905333477
// MI455X (gfx1250) — compile-verified, matrix operand order checked by reading, not run on hardware
//
#include <hip/hip_runtime.h>

#define S_LEN  2048
#define D_MOD  1024
#define N_HEAD 16
#define D_HEAD 64

typedef _Float16 f16;
typedef __attribute__((ext_vector_type(16))) _Float16 v16h;
typedef __attribute__((ext_vector_type(8)))  _Float16 v8h;
typedef __attribute__((ext_vector_type(8)))  float    v8f;

__device__ __forceinline__ v8f wmma_f16(v16h a, v16h b, v8f c) {
    // D = A(16x32 f16) * B(32x16 f16) + C(16x16 f32)
    return __builtin_amdgcn_wmma_f32_16x16x32_f16(false, a, false, b, (short)0, c,
                                                  false, false);
}

// 16-bit A/B fragment K-index mapping (ISA 7.12.2):
// lane group g (lane>>4): half h -> K = ((h<8)? h : h+8) + 8*g
__device__ __forceinline__ int kmap(int h, int g) {
    return ((h < 8) ? h : (h + 8)) + (g << 3);
}

// Fragment from a row-major f16 matrix: per-lane row, K contiguous along the row.
// Works for A (lane-row = M) and for B of x@W^T (lane-row = N, since B[k][n] = W[n][k]).
// Two contiguous 16-byte chunks -> two global_load_b128 per fragment.
__device__ __forceinline__ v16h frag_rm_f16(const f16* __restrict__ src, int ld,
                                            int row, int k0, int g) {
    v16h f;
    const f16* p = src + (size_t)row * ld + k0 + (g << 3);
#pragma unroll
    for (int h = 0; h < 8; ++h) f[h] = p[h];
#pragma unroll
    for (int h = 0; h < 8; ++h) f[h + 8] = p[h + 16];
    return f;
}

// B fragment whose K dimension walks the rows of a row-major matrix (used for V):
// B[k][n] = src[(k0+k)*ld + col]
__device__ __forceinline__ v16h frag_col_f16(const f16* __restrict__ src, int ld,
                                             int col, int k0, int g) {
    v16h f;
#pragma unroll
    for (int h = 0; h < 16; ++h) f[h] = src[(size_t)(k0 + kmap(h, g)) * ld + col];
    return f;
}

// One-shot fp32 -> f16 staging (vectorized: b128+b128 in, b128 out)
__global__ __launch_bounds__(256) void cvt_f32_f16_kernel(
    const float* __restrict__ src, f16* __restrict__ dst, int n)
{
    const int i = (blockIdx.x * 256 + threadIdx.x) * 8;
    if (i >= n) return;
    const float4 a = *(const float4*)(src + i);
    const float4 b = *(const float4*)(src + i + 4);
    v8h o;
    o[0] = (f16)a.x; o[1] = (f16)a.y; o[2] = (f16)a.z; o[3] = (f16)a.w;
    o[4] = (f16)b.x; o[5] = (f16)b.y; o[6] = (f16)b.z; o[7] = (f16)b.w;
    *(v8h*)(dst + i) = o;
}

// Y = X(f16,[M,1024]) @ W^T(f16,[1024,1024]) + bias.
// One wave computes a 32x64 tile: 2 A-frags x 4 B-frags = 8 WMMAs per 32-wide k-step.
// MODE 0: f16 head-split output [B,H,S,64], scaled (QKV projections)
// MODE 1: f32 flat output [M,1024]            (final projection)
template <int MODE>
__global__ __launch_bounds__(32) void gemm_kernel(
    const f16* __restrict__ X, const f16* __restrict__ W,
    const float* __restrict__ bias, f16* __restrict__ out16,
    float* __restrict__ out32, float scale)
{
    const int lane = threadIdx.x;
    const int g = lane >> 4, lr = lane & 15;
    const int m0 = blockIdx.x * 32;
    const int n0 = blockIdx.y * 64;
    v8f acc[2][4] = {};
    for (int k0 = 0; k0 < D_MOD; k0 += 32) {
        if (k0 + 32 < D_MOD) {   // global_prefetch_b8 of the next k-slice
            __builtin_prefetch(X + (size_t)(m0 + lr) * D_MOD + k0 + 32, 0, 1);
            __builtin_prefetch(W + (size_t)(n0 + lr) * D_MOD + k0 + 32, 0, 1);
        }
        const v16h a0 = frag_rm_f16(X, D_MOD, m0 + lr,      k0, g);
        const v16h a1 = frag_rm_f16(X, D_MOD, m0 + 16 + lr, k0, g);
#pragma unroll
        for (int t = 0; t < 4; ++t) {
            const v16h b = frag_rm_f16(W, D_MOD, n0 + t * 16 + lr, k0, g);
            acc[0][t] = wmma_f16(a0, b, acc[0][t]);
            acc[1][t] = wmma_f16(a1, b, acc[1][t]);
        }
    }
    // C layout: VGPR r, lanes 0-15 -> M=r, lanes 16-31 -> M=r+8; col = lane&15
#pragma unroll
    for (int u = 0; u < 2; ++u) {
#pragma unroll
        for (int t = 0; t < 4; ++t) {
            const int n = n0 + t * 16 + lr;
            const float bn = bias[n];
#pragma unroll
            for (int r = 0; r < 8; ++r) {
                const int m = m0 + u * 16 + r + 8 * g;
                if (MODE == 0) {
                    const int hh = n >> 6, dd = n & (D_HEAD - 1);
                    const int bidx = m >> 11, ss = m & (S_LEN - 1);
                    out16[(((size_t)bidx * N_HEAD + hh) * S_LEN + ss) * D_HEAD + dd] =
                        (f16)((acc[u][t][r] + bn) * scale);
                } else {
                    out32[(size_t)m * D_MOD + n] = acc[u][t][r] + bn;
                }
            }
        }
    }
}

// Flash attention, one wave per (16 query rows, head, batch). Causal.
__global__ __launch_bounds__(32) void flash_attn_kernel(
    const f16* __restrict__ Q, const f16* __restrict__ K,
    const f16* __restrict__ V, f16* __restrict__ ctx)
{
    __shared__ f16 pbuf[16 * 32];   // P tile staged C-layout -> A-layout
    const int lane = threadIdx.x;
    const int g = lane >> 4, lr = lane & 15;
    const int m0 = blockIdx.x * 16;
    const int hh = blockIdx.y, bb = blockIdx.z;
    const size_t head_off = ((size_t)bb * N_HEAD + hh) * S_LEN * D_HEAD;
    const f16* Qh = Q + head_off;
    const f16* Kh = K + head_off;
    const f16* Vh = V + head_off;

    // Q tile fragments (16 rows x 64 dims), loaded once
    const v16h aq0 = frag_rm_f16(Qh, D_HEAD, m0 + lr, 0, g);
    const v16h aq1 = frag_rm_f16(Qh, D_HEAD, m0 + lr, 32, g);

    float mrow[8], lsum[8];
    v8f acc[4] = {};
#pragma unroll
    for (int r = 0; r < 8; ++r) { mrow[r] = -1.0e30f; lsum[r] = 0.0f; }

    for (int j0 = 0; j0 <= m0 + 15; j0 += 32) {   // causal key-block loop
        v8f s[2] = {};
#pragma unroll
        for (int nf = 0; nf < 2; ++nf) {          // scores = Q @ K^T  (4 WMMAs)
            v16h bk = frag_rm_f16(Kh, D_HEAD, j0 + nf * 16 + lr, 0, g);
            s[nf] = wmma_f16(aq0, bk, s[nf]);
            bk = frag_rm_f16(Kh, D_HEAD, j0 + nf * 16 + lr, 32, g);
            s[nf] = wmma_f16(aq1, bk, s[nf]);
        }
        // causal mask: col key = j0 + nf*16 + (lane&15), row q = m0 + r + 8*g
#pragma unroll
        for (int nf = 0; nf < 2; ++nf) {
            const int key = j0 + nf * 16 + lr;
#pragma unroll
            for (int r = 0; r < 8; ++r) {
                const int qr = m0 + r + 8 * g;
                if (key > qr) s[nf][r] = -1.0e30f;
            }
        }
        // row max across the 16 lanes sharing each row (xor 1,2,4,8 stays in-group)
        float bm[8];
#pragma unroll
        for (int r = 0; r < 8; ++r) bm[r] = fmaxf(s[0][r], s[1][r]);
#pragma unroll
        for (int off = 1; off < 16; off <<= 1)
#pragma unroll
            for (int r = 0; r < 8; ++r)
                bm[r] = fmaxf(bm[r], __shfl_xor(bm[r], off, 32));
        // online softmax update
        float rs[8];
#pragma unroll
        for (int r = 0; r < 8; ++r) {
            const float mn   = fmaxf(mrow[r], bm[r]);
            const float corr = __expf(mrow[r] - mn);
            s[0][r] = __expf(s[0][r] - mn);
            s[1][r] = __expf(s[1][r] - mn);
            rs[r]   = s[0][r] + s[1][r];
            lsum[r] *= corr;
            mrow[r]  = mn;
#pragma unroll
            for (int t = 0; t < 4; ++t) acc[t][r] *= corr;
        }
#pragma unroll
        for (int off = 1; off < 16; off <<= 1)
#pragma unroll
            for (int r = 0; r < 8; ++r)
                rs[r] += __shfl_xor(rs[r], off, 32);
#pragma unroll
        for (int r = 0; r < 8; ++r) lsum[r] += rs[r];

        // P: C-layout -> LDS (row-major 16x32 f16) -> reload as A fragment
#pragma unroll
        for (int nf = 0; nf < 2; ++nf)
#pragma unroll
            for (int r = 0; r < 8; ++r)
                pbuf[(r + 8 * g) * 32 + nf * 16 + lr] = (f16)s[nf][r];
        __syncthreads();
        const v16h pf = frag_rm_f16(pbuf, 32, lr, 0, g);
#pragma unroll
        for (int t = 0; t < 4; ++t) {             // ctx += P @ V  (4 WMMAs)
            const v16h bv = frag_col_f16(Vh, D_HEAD, t * 16 + lr, j0, g);
            acc[t] = wmma_f16(pf, bv, acc[t]);
        }
        __syncthreads();
    }
    // normalize and store ctx in merged-head [B, S, H*64] order
#pragma unroll
    for (int t = 0; t < 4; ++t) {
#pragma unroll
        for (int r = 0; r < 8; ++r) {
            const int qr = m0 + r + 8 * g;
            ctx[((size_t)bb * S_LEN + qr) * D_MOD + hh * D_HEAD + t * 16 + lr] =
                (f16)(acc[t][r] / lsum[r]);
        }
    }
}

extern "C" void kernel_launch(void* const* d_in, const int* in_sizes, int n_in,
                              void* d_out, int out_size, void* d_ws, size_t ws_size,
                              hipStream_t stream) {
    const float* q  = (const float*)d_in[0];
    const float* k  = (const float*)d_in[1];
    const float* v  = (const float*)d_in[2];
    // d_in[3]: causal mask (int32) — causality is computed analytically
    const float* wq = (const float*)d_in[4];
    const float* bq = (const float*)d_in[5];
    const float* wk = (const float*)d_in[6];
    const float* bk = (const float*)d_in[7];
    const float* wv = (const float*)d_in[8];
    const float* bv = (const float*)d_in[9];
    const float* wo = (const float*)d_in[10];
    const float* bo = (const float*)d_in[11];

    const size_t NX = (size_t)2 * S_LEN * D_MOD;   // 4M elements (activations)
    const size_t NW = (size_t)D_MOD * D_MOD;       // 1M elements (weights)

    // Workspace map (f16 elements), 56 MB total:
    //   [0, 3*NX)          staged fp16 inputs  qx, kx, vx
    //   [3*NX, 3*NX+4*NW)  staged fp16 weights wq16, wk16, wv16, wo16
    //   [.., +3*NX)        head-split projections q16, k16, v16
    //   ctx16 aliases qx (the staged-Q input is dead after the Q projection)
    f16* qx   = (f16*)d_ws;
    f16* kx   = qx + NX;
    f16* vx   = kx + NX;
    f16* wq16 = vx + NX;
    f16* wk16 = wq16 + NW;
    f16* wv16 = wk16 + NW;
    f16* wo16 = wv16 + NW;
    f16* q16  = wo16 + NW;
    f16* k16  = q16 + NX;
    f16* v16  = k16 + NX;
    f16* c16  = qx;                                // alias (safe: see map)

    const int cvtBlk = 256;
    const int gX = (int)(NX / 8 / cvtBlk);         // 2048 blocks
    const int gW = (int)(NW / 8 / cvtBlk);         // 512 blocks
    cvt_f32_f16_kernel<<<gX, cvtBlk, 0, stream>>>(q,  qx,   (int)NX);
    cvt_f32_f16_kernel<<<gX, cvtBlk, 0, stream>>>(k,  kx,   (int)NX);
    cvt_f32_f16_kernel<<<gX, cvtBlk, 0, stream>>>(v,  vx,   (int)NX);
    cvt_f32_f16_kernel<<<gW, cvtBlk, 0, stream>>>(wq, wq16, (int)NW);
    cvt_f32_f16_kernel<<<gW, cvtBlk, 0, stream>>>(wk, wk16, (int)NW);
    cvt_f32_f16_kernel<<<gW, cvtBlk, 0, stream>>>(wv, wv16, (int)NW);
    cvt_f32_f16_kernel<<<gW, cvtBlk, 0, stream>>>(wo, wo16, (int)NW);

    const dim3 gGemm(2 * S_LEN / 32, D_MOD / 64);  // 128 x 16 tiles of 32x64
    // fold 1/sqrt(64) into the Q projection
    gemm_kernel<0><<<gGemm, 32, 0, stream>>>(qx, wq16, bq, q16, nullptr, 0.125f);
    gemm_kernel<0><<<gGemm, 32, 0, stream>>>(kx, wk16, bk, k16, nullptr, 1.0f);
    gemm_kernel<0><<<gGemm, 32, 0, stream>>>(vx, wv16, bv, v16, nullptr, 1.0f);

    const dim3 gAttn(S_LEN / 16, N_HEAD, 2);       // 128 x 16 x 2
    flash_attn_kernel<<<gAttn, 32, 0, stream>>>(q16, k16, v16, c16);

    gemm_kernel<1><<<gGemm, 32, 0, stream>>>(c16, wo16, bo, nullptr,
                                             (float*)d_out, 1.0f);
}
